// PAM_Module_v5_42872363549061
// MI455X (gfx1250) — matrix operand order repaired and checked by reading, not compiled, not run on hardware
//
#include <hip/hip_runtime.h>

// ---------------------------------------------------------------------------
// PAM module for MI455X (gfx1250): bf16 WMMA (v_wmma_f32_16x16x32_bf16),
// two-pass fused attention (no NxN matrix materialization), TDM async tensor
// loads (tensor_load_to_lds + s_wait_tensorcnt) double-buffering the Q chunk.
// B=4, C=512, H=W=64, N=4096, inter=64.
// ---------------------------------------------------------------------------

typedef __bf16 bf16;
typedef __attribute__((ext_vector_type(8)))  __bf16 v8bf;
typedef __attribute__((ext_vector_type(16))) __bf16 v16bf;
typedef __attribute__((ext_vector_type(8)))  float  v8f;
typedef __attribute__((ext_vector_type(4)))  float  v4f;
typedef __attribute__((ext_vector_type(4)))  unsigned v4u;
typedef __attribute__((ext_vector_type(4)))  int   v4i;
typedef __attribute__((ext_vector_type(8)))  int   v8i;

#define DEVI static __device__ __forceinline__

static constexpr int Bk = 4, Ck = 512, Nk = 4096, Dk = 64;

#if __has_builtin(__builtin_amdgcn_tensor_load_to_lds) && \
    __has_builtin(__builtin_amdgcn_s_wait_tensorcnt)
#define HAVE_TDM 1
#else
#define HAVE_TDM 0
#endif

DEVI bf16 f2bf(float f) { return (bf16)f; }

// A-operand (16 x K=32 bf16), source row-major, per-ISA layout:
// lane<16: halves 0..7 = K kb..kb+7, 8..15 = K kb+16..kb+23 ; lane>=16: +8
DEVI v16bf make_a_frag(const bf16* base, int ld, int row, int kbase, int lane) {
  const bf16* p = base + (size_t)row * ld + kbase + ((lane >> 4) << 3);
  v8bf lo = *(const v8bf*)p;
  v8bf hi = *(const v8bf*)(p + 16);
  v16bf a;
#pragma unroll
  for (int i = 0; i < 8; ++i) { a[i] = lo[i]; a[i + 8] = hi[i]; }
  return a;
}

// B-operand (K=32 x 16 bf16) from *transposed* storage (row = N-column,
// contiguous K): lane<16 holds K kb..kb+7, kb+16..kb+23 of col; lane>=16: +8.
DEVI v16bf make_b_fragT(const bf16* baseT, int ld, int col, int kbase, int lane) {
  const bf16* p = baseT + (size_t)col * ld + kbase + ((lane >> 4) << 3);
  v8bf lo = *(const v8bf*)p;
  v8bf hi = *(const v8bf*)(p + 16);
  v16bf b;
#pragma unroll
  for (int i = 0; i < 8; ++i) { b[i] = lo[i]; b[i + 8] = hi[i]; }
  return b;
}

DEVI v8f wmma_bf16(v16bf a, v16bf b, v8f c) {
  return __builtin_amdgcn_wmma_f32_16x16x32_bf16(false, a, false, b, (short)0, c,
                                                 false, false);
}

// branchless  -sqrt(max(d2,0))/N : v_max_num_f32 + v_sqrt_f32, no EXEC juggling
DEVI float neg_dist(float d2, float invN) {
  return -__builtin_amdgcn_sqrtf(fmaxf(d2, 0.0f)) * invN;
}

#if HAVE_TDM
// TDM D# for a 2D bf16 tile: 32 rows x 64 cols, row stride 64 elements,
// gsrc = tile start, lds_off = destination LDS byte offset.
DEVI void tdm_load_q32x64(const bf16* gsrc, unsigned lds_off) {
  unsigned long long ga = (unsigned long long)(const void*)gsrc;
  v4u g0;
  g0[0] = 1u;                   // count=1, user descriptor
  g0[1] = lds_off;              // lds_addr (bytes)
  g0[2] = (unsigned)ga;         // global_addr[31:0]
  g0[3] = (unsigned)((ga >> 32) & 0x1FFFFFFu) | (2u << 30);  // addr hi | type=2
  v8i g1;
  g1[0] = 0x00010000;           // workgroup_mask=0, data_size=1 (2 bytes)
  g1[1] = (int)(64u << 16);     // tensor_dim0 = 64   (bits 48..79)
  g1[2] = (int)(4096u << 16);   // tensor_dim1 = 4096 (bits 80..111)
  g1[3] = (int)(64u << 16);     // tile_dim0 = 64     (bits 112..127)
  g1[4] = 32;                   // tile_dim1 = 32, tile_dim2 = 0
  g1[5] = 64;                   // tensor_dim0_stride = 64 (bits 160..207)
  g1[6] = 0;
  g1[7] = 0;
  v4i z4 = {0, 0, 0, 0};
#if defined(__clang_major__) && __clang_major__ >= 23
  v8i z8 = {0, 0, 0, 0, 0, 0, 0, 0};
  __builtin_amdgcn_tensor_load_to_lds(g0, g1, z4, z4, z8, 0);
#else
  __builtin_amdgcn_tensor_load_to_lds(g0, g1, z4, z4, 0);
#endif
}
#endif

// ---------------------------------------------------------------------------
__global__ void k_f32_to_bf16(const float* __restrict__ in, bf16* __restrict__ out,
                              int n) {
  int i = (blockIdx.x * blockDim.x + threadIdx.x) * 8;
  if (i + 8 <= n) {
    v4f a = *(const v4f*)(in + i);
    v4f b = *(const v4f*)(in + i + 4);
    v8bf o;
#pragma unroll
    for (int e = 0; e < 4; ++e) { o[e] = f2bf(a[e]); o[e + 4] = f2bf(b[e]); }
    *(v8bf*)(out + i) = o;
  }
}

// x (B,C,N) f32 -> xT (B,N,C) bf16, tiled 32x32 transpose through LDS.
__global__ __launch_bounds__(256) void k_x_to_bf16T(const float* __restrict__ x,
                                                    bf16* __restrict__ xT) {
  __shared__ float tile[32][33];
  const int b = blockIdx.z;
  const int c0 = blockIdx.y * 32;
  const int n0 = blockIdx.x * 32;
  const int t0 = threadIdx.x & 31;
  const int t1 = threadIdx.x >> 5;  // 0..7
#pragma unroll
  for (int i = 0; i < 4; ++i) {
    int cl = t1 + i * 8;
    tile[cl][t0] = x[((size_t)b * Ck + c0 + cl) * Nk + n0 + t0];
  }
  __syncthreads();
#pragma unroll
  for (int i = 0; i < 4; ++i) {
    int nl = t1 + i * 8;
    xT[((size_t)b * Nk + n0 + nl) * Ck + c0 + t0] = f2bf(tile[t0][nl]);
  }
}

// ---------------------------------------------------------------------------
// C[M x 4096] = W[M x 512] @ X[512 x 4096] + bias ; per batch (blockIdx.z).
// A-frags from W rows, B-frags from XT rows — both direct from global, no LDS.
// MODE 0: V conv -> write (b, ci, hw) directly.
// MODE 1: Q/K conv -> write scrambled (b, n=ci*64+(hw>>6), d=hw&63).
template <int MODE>
__global__ __launch_bounds__(256) void k_gemm_wx(const bf16* __restrict__ W,
                                                 const bf16* __restrict__ XT,
                                                 const float* __restrict__ bias,
                                                 bf16* __restrict__ out) {
  const int lane = threadIdx.x & 31;
  const int w = threadIdx.x >> 5;
  const int nblock = blockIdx.x * 128;
  const int mbase = blockIdx.y * 64;
  const int b = blockIdx.z;
  const bf16* XTb = XT + (size_t)b * Nk * Ck;
  const int mw = (w & 3) * 16;
  const int nw = (w >> 2) * 64;
  const int half = lane >> 4;

  v8f acc[4];
#pragma unroll
  for (int j = 0; j < 4; ++j) acc[j] = 0.0f;

  for (int kk = 0; kk < Ck; kk += 32) {
    v16bf a = make_a_frag(W, Ck, mbase + mw + (lane & 15), kk, lane);
#pragma unroll
    for (int j = 0; j < 4; ++j) {
      v16bf bf =
          make_b_fragT(XTb, Ck, nblock + nw + j * 16 + (lane & 15), kk, lane);
      acc[j] = wmma_bf16(a, bf, acc[j]);
    }
  }
#pragma unroll
  for (int j = 0; j < 4; ++j) {
#pragma unroll
    for (int r = 0; r < 8; ++r) {
      int ci = mbase + mw + r + 8 * half;
      int hw = nblock + nw + j * 16 + (lane & 15);
      float v = acc[j][r] + bias[ci];
      if (MODE == 0) {
        out[((size_t)b * Ck + ci) * Nk + hw] = f2bf(v);
      } else {
        int n = ci * 64 + (hw >> 6);
        int d = hw & 63;
        out[((size_t)b * Nk + n) * 64 + d] = f2bf(v);
      }
    }
  }
}

// ---------------------------------------------------------------------------
__global__ void k_norm(const bf16* __restrict__ Q, float* __restrict__ out,
                       int rows) {
  int r = blockIdx.x * blockDim.x + threadIdx.x;
  if (r >= rows) return;
  const bf16* p = Q + (size_t)r * Dk;
  float s = 0.f;
#pragma unroll
  for (int i = 0; i < 8; ++i) {
    v8bf v = *(const v8bf*)(p + i * 8);
#pragma unroll
    for (int e = 0; e < 8; ++e) { float f = (float)v[e]; s += f * f; }
  }
  out[r] = s;
}

// ---------------------------------------------------------------------------
// Pass A: inv_se[b,n] = 1 / sum_m exp(-sqrt(max(|q_n|^2+|k_m|^2-2 q.k,0))/N)
__global__ __launch_bounds__(256) void k_row_denom(
    const bf16* __restrict__ Q, const bf16* __restrict__ K,
    const float* __restrict__ qn2, const float* __restrict__ kn2,
    float* __restrict__ inv_se) {
  const int lane = threadIdx.x & 31;
  const int w = threadIdx.x >> 5;
  const int b = blockIdx.y;
  const int nbase = blockIdx.x * 128 + w * 16;
  const bf16* Qb = Q + (size_t)b * Nk * Dk;
  const bf16* Kb = K + (size_t)b * Nk * Dk;
  const float* qn2b = qn2 + b * Nk;
  const float* kn2b = kn2 + b * Nk;
  const int half = lane >> 4;
  const float invN = 1.0f / (float)Nk;

  v16bf a0 = make_a_frag(Qb, Dk, nbase + (lane & 15), 0, lane);
  v16bf a1 = make_a_frag(Qb, Dk, nbase + (lane & 15), 32, lane);
  float qrow[8], sum[8];
#pragma unroll
  for (int r = 0; r < 8; ++r) {
    qrow[r] = qn2b[nbase + r + 8 * half];
    sum[r] = 0.f;
  }
  for (int mt = 0; mt < Nk / 16; ++mt) {
    int mbase = mt * 16;
    v16bf b0 = make_b_fragT(Kb, Dk, mbase + (lane & 15), 0, lane);
    v16bf b1 = make_b_fragT(Kb, Dk, mbase + (lane & 15), 32, lane);
    v8f c = 0.0f;
    c = wmma_bf16(a0, b0, c);
    c = wmma_bf16(a1, b1, c);
    float kcol = kn2b[mbase + (lane & 15)];
#pragma unroll
    for (int r = 0; r < 8; ++r) {
      float d2 = qrow[r] + kcol - 2.0f * c[r];
      sum[r] += __expf(neg_dist(d2, invN));
    }
  }
#pragma unroll
  for (int r = 0; r < 8; ++r) {
#pragma unroll
    for (int m = 1; m < 16; m <<= 1) sum[r] += __shfl_xor(sum[r], m, 32);
    if ((lane & 15) == 0) inv_se[b * Nk + nbase + r + 8 * half] = 1.0f / sum[r];
  }
}

// ---------------------------------------------------------------------------
// Pass B: out[b,c,m] = gamma * sum_n V[c,n]*attn[n,m] + x[b,c,m]
// Block 256 thr (8 waves): c-block 256 (32/wave), m-block 64. Loop n in
// chunks of 32: TDM double-buffers the 32x64 Q chunk into LDS; phase1 computes
// the attn chunk into LDS (B-operand layout); phase2 runs V x attn on WMMA.
__global__ __launch_bounds__(256) void k_attn_out(
    const bf16* __restrict__ Q, const bf16* __restrict__ K,
    const bf16* __restrict__ V, const float* __restrict__ qn2,
    const float* __restrict__ kn2, const float* __restrict__ inv_se,
    const float* __restrict__ x, const float* __restrict__ gamma,
    float* __restrict__ out) {
  __shared__ __align__(16) bf16 attn_s[64 * 40];  // [m-local][n-local(32)+pad]
#if HAVE_TDM
  __shared__ __align__(16) bf16 q_s[2][32 * 64];  // double-buffered Q chunk
#endif
  const int lane = threadIdx.x & 31;
  const int w = threadIdx.x >> 5;
  const int mblock = blockIdx.x * 64;
  const int cblock = blockIdx.y * 256;
  const int b = blockIdx.z;
  const bf16* Qb = Q + (size_t)b * Nk * Dk;
  const bf16* Kb = K + (size_t)b * Nk * Dk;
  const bf16* Vb = V + (size_t)b * Ck * Nk;
  const float* qn2b = qn2 + b * Nk;
  const float* kn2b = kn2 + b * Nk;
  const float* iseb = inv_se + b * Nk;
  const int half = lane >> 4;
  const float invN = 1.0f / (float)Nk;

  // phase-1 tile per wave: n-sub {0,16}, m-sub {0,16,32,48}
  const int nsub = (w & 1) * 16;
  const int msub = (w >> 1) * 16;
  // loop-invariant K-side operands (fixed m columns)
  v16bf kb0 = make_b_fragT(Kb, Dk, mblock + msub + (lane & 15), 0, lane);
  v16bf kb1 = make_b_fragT(Kb, Dk, mblock + msub + (lane & 15), 32, lane);
  const float kcol = kn2b[mblock + msub + (lane & 15)];

  const int csub = cblock + w * 32;
  v8f acc[2][4];
#pragma unroll
  for (int ct = 0; ct < 2; ++ct)
#pragma unroll
    for (int j = 0; j < 4; ++j) acc[ct][j] = 0.0f;

#if HAVE_TDM
  const unsigned qs_off0 = (unsigned)(unsigned long long)(const void*)&q_s[0][0];
  const unsigned qs_off1 = (unsigned)(unsigned long long)(const void*)&q_s[1][0];
  if (w == 0) tdm_load_q32x64(Qb, qs_off0);  // preload chunk 0 into buf 0
#endif

  for (int nc = 0; nc < Nk; nc += 32) {
    const int cur = (nc >> 5) & 1;
#if HAVE_TDM
    if (w == 0) {
      // kick off next chunk's DMA (wraps harmlessly on the last iteration),
      // then wait until the current chunk (issued one iteration ago) landed.
      int nnext = (nc + 32 < Nk) ? nc + 32 : 0;
      tdm_load_q32x64(Qb + (size_t)nnext * Dk, cur ? qs_off0 : qs_off1);
      __builtin_amdgcn_s_wait_tensorcnt(1);
    }
#endif
    __syncthreads();  // q_s[cur] ready; attn_s from last iter fully consumed

    // ---- phase 1: attn chunk -> LDS
#if HAVE_TDM
    const bf16* qsrc = &q_s[cur][0];
    v16bf qa0 = make_a_frag(qsrc, Dk, nsub + (lane & 15), 0, lane);
    v16bf qa1 = make_a_frag(qsrc, Dk, nsub + (lane & 15), 32, lane);
#else
    v16bf qa0 = make_a_frag(Qb, Dk, nc + nsub + (lane & 15), 0, lane);
    v16bf qa1 = make_a_frag(Qb, Dk, nc + nsub + (lane & 15), 32, lane);
#endif
    v8f s = 0.0f;
    s = wmma_bf16(qa0, kb0, s);
    s = wmma_bf16(qa1, kb1, s);
    v8bf pk;
#pragma unroll
    for (int r = 0; r < 8; ++r) {
      int n = nc + nsub + 8 * half + r;
      float d2 = qn2b[n] + kcol - 2.0f * s[r];
      pk[r] = f2bf(__expf(neg_dist(d2, invN)) * iseb[n]);
    }
    *(v8bf*)&attn_s[(msub + (lane & 15)) * 40 + nsub + 8 * half] = pk;
    __syncthreads();

    // ---- phase 2: out += V[:,nc..nc+32] @ attn_chunk
    __builtin_prefetch(Vb + (size_t)(csub + (lane & 15)) * Nk + nc + 32, 0, 1);
    v16bf av0 = make_a_frag(Vb, Nk, csub + (lane & 15), nc, lane);
    v16bf av1 = make_a_frag(Vb, Nk, csub + 16 + (lane & 15), nc, lane);
#pragma unroll
    for (int j = 0; j < 4; ++j) {
      v16bf ab = make_b_fragT(attn_s, 40, j * 16 + (lane & 15), 0, lane);
      acc[0][j] = wmma_bf16(av0, ab, acc[0][j]);
      acc[1][j] = wmma_bf16(av1, ab, acc[1][j]);
    }
    // loop-top barrier orders attn_s WAR for the next iteration
  }
  const float g = gamma[0];
#pragma unroll
  for (int ct = 0; ct < 2; ++ct)
#pragma unroll
    for (int j = 0; j < 4; ++j)
#pragma unroll
      for (int r = 0; r < 8; ++r) {
        int c = csub + ct * 16 + r + 8 * half;
        int m = mblock + j * 16 + (lane & 15);
        size_t idx = ((size_t)b * Ck + c) * Nk + m;
        out[idx] = g * acc[ct][j][r] + x[idx];
      }
}

// ---------------------------------------------------------------------------
extern "C" void kernel_launch(void* const* d_in, const int* in_sizes, int n_in,
                              void* d_out, int out_size, void* d_ws,
                              size_t ws_size, hipStream_t stream) {
  const float* x = (const float*)d_in[0];
  const float* wq = (const float*)d_in[1];
  const float* bq = (const float*)d_in[2];
  const float* wk = (const float*)d_in[3];
  const float* bk = (const float*)d_in[4];
  const float* wv = (const float*)d_in[5];
  const float* bv = (const float*)d_in[6];
  const float* gamma = (const float*)d_in[7];
  float* out = (float*)d_out;

  char* ws = (char*)d_ws;
  size_t off = 0;
  auto carve = [&](size_t bytes) {
    void* p = ws + off;
    off += (bytes + 255) & ~(size_t)255;
    return p;
  };
  bf16* XT  = (bf16*)carve((size_t)Bk * Nk * Ck * 2);   // 16 MB, (B,N,C)
  bf16* Wqb = (bf16*)carve((size_t)Dk * Ck * 2);
  bf16* Wkb = (bf16*)carve((size_t)Dk * Ck * 2);
  bf16* Wvb = (bf16*)carve((size_t)Ck * Ck * 2);
  bf16* Qbf = (bf16*)carve((size_t)Bk * Nk * Dk * 2);   // 2 MB, (B,N,64)
  bf16* Kbf = (bf16*)carve((size_t)Bk * Nk * Dk * 2);
  bf16* Vbf = (bf16*)carve((size_t)Bk * Ck * Nk * 2);   // 16 MB, (B,C,N)
  float* qn2 = (float*)carve((size_t)Bk * Nk * 4);
  float* kn2 = (float*)carve((size_t)Bk * Nk * 4);
  float* ise = (float*)carve((size_t)Bk * Nk * 4);

  // 0) precision converts (x also transposed to (B,N,C))
  k_x_to_bf16T<<<dim3(Nk / 32, Ck / 32, Bk), 256, 0, stream>>>(x, XT);
  {
    int n = Dk * Ck;
    k_f32_to_bf16<<<n / (256 * 8), 256, 0, stream>>>(wq, Wqb, n);
    k_f32_to_bf16<<<n / (256 * 8), 256, 0, stream>>>(wk, Wkb, n);
    n = Ck * Ck;
    k_f32_to_bf16<<<n / (256 * 8), 256, 0, stream>>>(wv, Wvb, n);
  }
  // 1) 1x1 convs as WMMA GEMMs (LDS-free)
  k_gemm_wx<1><<<dim3(Nk / 128, Dk / 64, Bk), 256, 0, stream>>>(Wqb, XT, bq, Qbf);
  k_gemm_wx<1><<<dim3(Nk / 128, Dk / 64, Bk), 256, 0, stream>>>(Wkb, XT, bk, Kbf);
  k_gemm_wx<0><<<dim3(Nk / 128, Ck / 64, Bk), 256, 0, stream>>>(Wvb, XT, bv, Vbf);
  // 2) squared norms
  k_norm<<<(Bk * Nk) / 256, 256, 0, stream>>>(Qbf, qn2, Bk * Nk);
  k_norm<<<(Bk * Nk) / 256, 256, 0, stream>>>(Kbf, kn2, Bk * Nk);
  // 3) softmax denominators (pass A)
  k_row_denom<<<dim3(Nk / 128, Bk), 256, 0, stream>>>(Qbf, Kbf, qn2, kn2, ise);
  // 4) fused attn x V + residual (pass B)
  k_attn_out<<<dim3(Nk / 64, Ck / 256, Bk), 256, 0, stream>>>(
      Qbf, Kbf, Vbf, qn2, kn2, ise, x, gamma, out);
}
